// LossMeanCov_81612968558870
// MI455X (gfx1250) — matrix operand order repaired and checked by reading, not compiled, not run on hardware
//
#include <hip/hip_runtime.h>
#include <hip/hip_bf16.h>

typedef __attribute__((ext_vector_type(16))) _Float16 v16h;
typedef __attribute__((ext_vector_type(8)))  float    v8f;
typedef __attribute__((ext_vector_type(2)))  float    v2f;

#define NPTS 65536
#define DIMD 64
#define KCL  512

// ---------------------------------------------------------------------------
// Kernel 0: c2[k] = ||center_k||^2 in fp32
// ---------------------------------------------------------------------------
__global__ __launch_bounds__(256)
void cl_c2_kernel(const float* __restrict__ centers, float* __restrict__ c2) {
  int kk = blockIdx.x * blockDim.x + threadIdx.x;
  if (kk < KCL) {
    const float* c = centers + (size_t)kk * DIMD;
    float s = 0.f;
    for (int j = 0; j < DIMD; ++j) s += c[j] * c[j];
    c2[kk] = s;
  }
}

// ---------------------------------------------------------------------------
// Kernel 1: assignment.  One wave = 16 rows of x, full sweep over 512 cols.
// d2 = x2 - 2*(x@C^T) + c2 ; GEMM via v_wmma_f32_16x16x32_f16 (2 per 16x16
// tile, K=64).  Centers staged as f16 in LDS.  Deterministic argmin with
// first-index tie-break.
// ---------------------------------------------------------------------------
__global__ __launch_bounds__(256)
void cl_assign_kernel(const float* __restrict__ x,
                      const float* __restrict__ centers,
                      const float* __restrict__ c2,
                      int* __restrict__ pred) {
  __shared__ __attribute__((aligned(64))) _Float16 sC[KCL * DIMD];
  const int tid = threadIdx.x;
  for (int e = tid; e < KCL * DIMD; e += 256) sC[e] = (_Float16)centers[e];
  __syncthreads();

  const int wave = tid >> 5;
  const int lane = tid & 31;
  const int g = lane >> 4;   // half-wave index
  const int m = lane & 15;   // row within A tile / col within B,D tile
  const int rowbase = blockIdx.x * 128 + wave * 16;

  // A fragments (16x32 f16, two of them for K=0..31 and K=32..63).
  // Contiguous layout: half h (0..7) of frag0 holds K = g*16 + h, half h+8
  // holds K = g*16 + 8 + h  (row = m).
  const float* xr = x + (size_t)(rowbase + m) * DIMD;
  v16h a0, a1;
  float x2part = 0.f;
  for (int h = 0; h < 8; ++h) {
    float f0 = xr[g * 16 + h];
    float f1 = xr[g * 16 + 8 + h];
    float f2 = xr[32 + g * 16 + h];
    float f3 = xr[32 + g * 16 + 8 + h];
    a0[h]     = (_Float16)f0;
    a0[h + 8] = (_Float16)f1;
    a1[h]     = (_Float16)f2;
    a1[h + 8] = (_Float16)f3;
    x2part += f0 * f0 + f1 * f1 + f2 * f2 + f3 * f3;
  }
  // lane covers 32 of 64 dims of its row; complementary half is in lane^16
  float x2full = x2part + __shfl_xor(x2part, 16, 32);
  // lane needs x2 of rows g*8 + r (held by lane g*8+r, 0..15)
  float x2r[8];
  for (int r = 0; r < 8; ++r) x2r[r] = __shfl(x2full, g * 8 + r, 32);

  float bestv[8];
  int   besti[8];
  for (int r = 0; r < 8; ++r) { bestv[r] = 3.4e38f; besti[r] = 0; }

  for (int ctile = 0; ctile < KCL / 16; ++ctile) {
    const int colbase = ctile * 16;
    // B fragment (32x16 f16): lane = col m; halves 0..15 = K g*16 .. g*16+15
    const v16h* cpv = (const v16h*)(sC + (size_t)(colbase + m) * DIMD);
    v16h b0 = cpv[g];       // K 0..31   (this half-wave's 16)
    v16h b1 = cpv[2 + g];   // K 32..63
    v8f acc = {0.f, 0.f, 0.f, 0.f, 0.f, 0.f, 0.f, 0.f};
    acc = __builtin_amdgcn_wmma_f32_16x16x32_f16(false, a0, false, b0,
                                                 (short)0, acc, false, false);
    acc = __builtin_amdgcn_wmma_f32_16x16x32_f16(false, a1, false, b1,
                                                 (short)0, acc, false, false);
    const float c2n = c2[colbase + m];
    const int   idx = colbase + m;
    for (int r = 0; r < 8; ++r) {
      float v = x2r[r] + c2n - 2.0f * acc[r];   // D row = g*8 + r, col = m
      if (v < bestv[r]) { bestv[r] = v; besti[r] = idx; }
    }
  }
  // reduce argmin across the 16 lanes of each half-wave (first-index ties)
  for (int r = 0; r < 8; ++r) {
    float bv = bestv[r];
    int   bi = besti[r];
    for (int off = 8; off; off >>= 1) {
      float ov = __shfl_xor(bv, off, 32);
      int   oi = __shfl_xor(bi, off, 32);
      if (ov < bv || (ov == bv && oi < bi)) { bv = ov; bi = oi; }
    }
    if (m == 0) pred[rowbase + g * 8 + r] = bi;
  }
}

// ---------------------------------------------------------------------------
// Kernel 2: per-cluster statistics + per-cluster loss partial.
// One block per cluster; each wave scans a fixed 1/8 slice of pred (ballot),
// gathers matching rows in batches of 4 and accumulates X^T X (64x64) with
// 16 v_wmma_f32_16x16x4_f32 accumulators.  All reductions fixed-order =>
// deterministic.
// ---------------------------------------------------------------------------
__global__ __launch_bounds__(256)
void cl_stats_kernel(const float* __restrict__ x,
                     const int* __restrict__ pred,
                     const float* __restrict__ centers,
                     const float* __restrict__ ft,
                     const float* __restrict__ mt,
                     const float* __restrict__ ct,
                     float* __restrict__ partial) {
  const int k    = blockIdx.x;
  const int tid  = threadIdx.x;
  const int w    = tid >> 5;
  const int lane = tid & 31;
  const int g    = lane >> 4;
  const int m    = lane & 15;

  __shared__ __attribute__((aligned(16))) float red[4][DIMD * DIMD]; // 64KB
  __shared__ float ssum[8][DIMD];
  __shared__ int   scnt[8];
  __shared__ float smean[DIMD];
  __shared__ float spart[256];

  v8f acc[16];
  for (int i = 0; i < 16; ++i)
    for (int r = 0; r < 8; ++r) acc[i][r] = 0.f;
  v2f sac[4];
  for (int t = 0; t < 4; ++t) { sac[t].x = 0.f; sac[t].y = 0.f; }
  int cnt = 0;

  const int seg = NPTS / 8;
  const int beg = w * seg;
  for (int c0 = 0; c0 < seg; c0 += 32) {
    __builtin_prefetch(pred + beg + c0 + 256, 0, 1);   // global_prefetch_b8
    const int i = beg + c0 + lane;
    const bool match = (pred[i] == k);
    unsigned mask = __builtin_amdgcn_ballot_w32(match);
    cnt += __builtin_popcount(mask);
    while (mask) {                 // wave-uniform loop: EXEC stays all-1s
      int pbits[4];
      for (int j = 0; j < 4; ++j) {
        if (mask) {
          int b = __builtin_ctz(mask);
          pbits[j] = beg + c0 + b;
          mask &= mask - 1;
        } else {
          pbits[j] = -1;
        }
      }
      // f32 16x4 A / 4x16 B fragments: half-wave 0 carries K=0,1 (pts 0,1),
      // half-wave 1 carries K=2,3 (pts 2,3); A and B fragments coincide.
      const int p0i = g ? pbits[2] : pbits[0];
      const int p1i = g ? pbits[3] : pbits[1];
      v2f fr[4];
      for (int t = 0; t < 4; ++t) {
        float fx = (p0i >= 0) ? x[(size_t)p0i * DIMD + t * 16 + m] : 0.f;
        float fy = (p1i >= 0) ? x[(size_t)p1i * DIMD + t * 16 + m] : 0.f;
        fr[t].x = fx;
        fr[t].y = fy;
        sac[t].x += fx;            // per-lane partial of column sums
        sac[t].y += fy;
      }
      for (int di = 0; di < 4; ++di)
        for (int dj = 0; dj < 4; ++dj)
          acc[di * 4 + dj] = __builtin_amdgcn_wmma_f32_16x16x4_f32(
              false, fr[di], false, fr[dj], (short)0, acc[di * 4 + dj],
              false, false);
    }
  }

  // per-wave column sums: dim a = t*16 + m, halves combined via shfl_xor 16
  for (int t = 0; t < 4; ++t) {
    float s = sac[t].x + sac[t].y;
    s += __shfl_xor(s, 16, 32);
    if (g == 0) ssum[w][t * 16 + m] = s;
  }
  if (lane == 0) scnt[w] = cnt;

  // Gram reduction across waves: waves 0-3 write, waves 4-7 add, then 4->1.
  if (w < 4) {
    for (int di = 0; di < 4; ++di)
      for (int dj = 0; dj < 4; ++dj)
        for (int r = 0; r < 8; ++r) {
          int a = di * 16 + g * 8 + r;   // D row
          int b = dj * 16 + m;           // D col
          red[w][a * DIMD + b] = acc[di * 4 + dj][r];
        }
  }
  __syncthreads();
  if (w >= 4) {
    for (int di = 0; di < 4; ++di)
      for (int dj = 0; dj < 4; ++dj)
        for (int r = 0; r < 8; ++r) {
          int a = di * 16 + g * 8 + r;
          int b = dj * 16 + m;
          red[w - 4][a * DIMD + b] += acc[di * 4 + dj][r];
        }
  }
  __syncthreads();

  int nk = 0;
  for (int j = 0; j < 8; ++j) nk += scnt[j];
  if (tid < DIMD) {
    float s = 0.f;
    for (int j = 0; j < 8; ++j) s += ssum[j][tid];
    smean[tid] = (nk > 0) ? (s / (float)nk)
                          : centers[(size_t)k * DIMD + tid];
  }
  __syncthreads();

  const float fn      = (float)nk;
  const float denom   = (float)((nk - 1) > 1 ? (nk - 1) : 1);
  const float inv_cdd = 1.f / ((float)KCL * (float)DIMD * (float)DIMD);
  const float inv_cd  = 1.f / ((float)KCL * (float)DIMD);
  const float* ctk = ct + (size_t)k * DIMD * DIMD;
  float psum = 0.f;
  for (int e = tid * 16; e < tid * 16 + 16; ++e) {
    int a = e >> 6, b = e & 63;
    float S = red[0][e] + red[1][e] + red[2][e] + red[3][e];
    float cov = (nk > 1) ? (S - fn * smean[a] * smean[b]) / denom : 0.f;
    float d = cov - ctk[e];
    psum += d * d * inv_cdd;
  }
  if (tid < DIMD) {
    float dm = smean[tid] - mt[(size_t)k * DIMD + tid];
    psum += dm * dm * inv_cd;
  }
  if (tid == 0) {
    float dfil = fn / (float)NPTS - ft[k];
    psum += dfil * dfil / (float)KCL;
  }
  spart[tid] = psum;
  __syncthreads();
  if (tid == 0) {
    float s = 0.f;
    for (int j = 0; j < 256; ++j) s += spart[j];
    partial[k] = s;
  }
}

// ---------------------------------------------------------------------------
// Kernel 3: deterministic final sum of 512 per-cluster partials.
// ---------------------------------------------------------------------------
__global__ void cl_final_kernel(const float* __restrict__ partial,
                                float* __restrict__ out) {
  if (threadIdx.x == 0 && blockIdx.x == 0) {
    float s = 0.f;
    for (int i = 0; i < KCL; ++i) s += partial[i];
    out[0] = s;
  }
}

// ---------------------------------------------------------------------------
extern "C" void kernel_launch(void* const* d_in, const int* in_sizes, int n_in,
                              void* d_out, int out_size, void* d_ws,
                              size_t ws_size, hipStream_t stream) {
  (void)in_sizes; (void)n_in; (void)out_size; (void)ws_size;
  const float* x       = (const float*)d_in[0];  // [65536,64]
  const float* centers = (const float*)d_in[1];  // [512,64]
  const float* ftarget = (const float*)d_in[2];  // [512]
  const float* mtarget = (const float*)d_in[3];  // [512,64]
  const float* ctarget = (const float*)d_in[4];  // [512,64,64]
  float* out = (float*)d_out;

  char* ws = (char*)d_ws;
  int*   pred    = (int*)ws;                                   // 256 KB
  float* c2      = (float*)(ws + (size_t)NPTS * sizeof(int));  // 2 KB
  float* partial = c2 + KCL;                                   // 2 KB

  cl_c2_kernel<<<2, 256, 0, stream>>>(centers, c2);
  cl_assign_kernel<<<NPTS / 128, 256, 0, stream>>>(x, centers, c2, pred);
  cl_stats_kernel<<<KCL, 256, 0, stream>>>(x, pred, centers, ftarget,
                                           mtarget, ctarget, partial);
  cl_final_kernel<<<1, 32, 0, stream>>>(partial, out);
}
